// PhotonNet_21887153340473
// MI455X (gfx1250) — matrix operand order repaired and checked by reading, not compiled, not run on hardware
//
#include <hip/hip_runtime.h>
#include <hip/hip_bf16.h>

typedef __attribute__((ext_vector_type(16))) _Float16 v16h;
typedef __attribute__((ext_vector_type(8)))  _Float16 v8h;
typedef __attribute__((ext_vector_type(8)))  float    v8f;
typedef __attribute__((ext_vector_type(4)))  unsigned int u32x4;
typedef __attribute__((ext_vector_type(8)))  int      i32x8;
typedef __attribute__((ext_vector_type(4)))  int      i32x4;

#define BB     16
#define NN     2048
#define NH     (BB*NN)      // 32768 hits
#define IN_D   20
#define GRAV   64
#define HIDD   256
#define SPACE_D 16
#define PROP_D 64
#define PREHID 128
#define KNN    6

// ---------------------------------------------------------------------------
// Stage 1: per-event stats of raw input (mean/min/max over N=2048, F=20)
// ---------------------------------------------------------------------------
__global__ void in_stats_kernel(const float* __restrict__ x, float* __restrict__ stats) {
    int e = blockIdx.x;
    int f = threadIdx.x;
    if (f >= IN_D) return;
    const float* p = x + (size_t)e * NN * IN_D + f;
    float sum = 0.f, mn = 3.0e38f, mx = -3.0e38f;
    for (int n = 0; n < NN; ++n) {
        float v = p[(size_t)n * IN_D];
        sum += v; mn = fminf(mn, v); mx = fmaxf(mx, v);
    }
    stats[e * 60 + f]      = sum * (1.f / NN);
    stats[e * 60 + 20 + f] = mn;
    stats[e * 60 + 40 + f] = mx;
}

__global__ void build_x96_kernel(const float* __restrict__ x, const float* __restrict__ stats,
                                 _Float16* __restrict__ X96) {
    int t = blockIdx.x * blockDim.x + threadIdx.x;
    if (t >= NH * 96) return;
    int hit = t / 96, c = t % 96;
    int e = hit >> 11;
    float v;
    if (c < 60)       v = stats[e * 60 + c];
    else if (c < 80)  v = x[(size_t)hit * IN_D + (c - 60)];
    else              v = 0.f;
    X96[t] = (_Float16)v;
}

// ---------------------------------------------------------------------------
// Weight prep: fp32 (K x N) row-major -> f16 W^T (N x Kpad), zero-padded K
// ---------------------------------------------------------------------------
__global__ void prep_wt_kernel(const float* __restrict__ W, _Float16* __restrict__ Wt,
                               int Kdim, int Ndim, int Kpad) {
    int t = blockIdx.x * blockDim.x + threadIdx.x;
    if (t >= Ndim * Kpad) return;
    int n = t / Kpad, k = t % Kpad;
    Wt[t] = (_Float16)(k < Kdim ? W[(size_t)k * Ndim + n] : 0.f);
}

// ---------------------------------------------------------------------------
// Fragment helpers (CDNA5 wave32 WMMA layouts)
// ---------------------------------------------------------------------------
__device__ __forceinline__ v16h fragA(const _Float16* p) {   // p at (row, khalf*8)
    v8h lo = *(const v8h*)p;
    v8h hi = *(const v8h*)(p + 16);
    v16h r;
#pragma unroll
    for (int i = 0; i < 8; ++i) { r[i] = lo[i]; r[8 + i] = hi[i]; }
    return r;
}
__device__ __forceinline__ v16h fragB(const _Float16* p) {   // p at (col, khalf*8)
    v8h lo = *(const v8h*)p;
    v8h hi = *(const v8h*)(p + 16);
    v16h r;
#pragma unroll
    for (int i = 0; i < 8; ++i) { r[i] = lo[i]; r[8 + i] = hi[i]; }
    return r;
}

// ---------------------------------------------------------------------------
// TDM: stage one contiguous slab (elems f16) from global into LDS.
// D# built per CDNA5 ISA 8.3/8.4: Group0 {count=1, lds_addr, global_addr,
// type=2}; Group1 {data_size=2B, tensor_dim0=elems, tensor_dim1=1,
// tile_dim0=elems, tile_dim1=1, stride0=elems}. Groups 2/3 zero (2D tensor).
// ---------------------------------------------------------------------------
__device__ __forceinline__ void tdm_load_slab_f16(const _Float16* gsrc, unsigned lds_off, int elems) {
    unsigned long long ga = (unsigned long long)(uintptr_t)gsrc;
    u32x4 g0;
    g0[0] = 1u;                                             // count=1 (valid)
    g0[1] = lds_off;                                        // lds_addr
    g0[2] = (unsigned)(ga & 0xFFFFFFFFu);                   // global_addr[31:0]
    g0[3] = (unsigned)((ga >> 32) & 0x01FFFFFFu) | 0x80000000u; // [56:32] | type=2
    i32x8 g1;
    g1[0] = 0x00010000;                                     // data_size=1 (2 bytes)
    g1[1] = (elems & 0xFFFF) << 16;                         // tensor_dim0[15:0]
    g1[2] = ((elems >> 16) & 0xFFFF) | (1 << 16);           // td0[31:16] | tensor_dim1=1
    g1[3] = (elems & 0xFFFF) << 16;                         // tile_dim0 = elems
    g1[4] = 1;                                              // tile_dim1=1, tile_dim2=0
    g1[5] = elems;                                          // tensor_dim0_stride lo
    g1[6] = 0;
    g1[7] = 0;
    i32x4 z = {0, 0, 0, 0};
#if __clang_major__ >= 23
    i32x8 z8 = {0, 0, 0, 0, 0, 0, 0, 0};
    __builtin_amdgcn_tensor_load_to_lds(g0, g1, z, z, z8, 0);
#else
    __builtin_amdgcn_tensor_load_to_lds(g0, g1, z, z, 0);
#endif
}

// ---------------------------------------------------------------------------
// WMMA GEMM, LDS/TDM-staged B panel, 2x2 register blocking.
// Block = 8 waves; block covers 256 rows x 32 cols; wave owns 32x32.
// Requires M%256==0, Ncols%32==0, Kpad<=128, ldw==Kpad.
// ---------------------------------------------------------------------------
__global__ void gemm_f16_wmma_lds_kernel(const _Float16* __restrict__ A, int lda,
                                         const _Float16* __restrict__ Wt, int ldw,
                                         const float* __restrict__ bias,
                                         const float* __restrict__ Cin,
                                         float* __restrict__ C,
                                         int M, int Ncols, int Kpad, int relu) {
    __shared__ _Float16 ldsB[32 * 128];

    const int wid  = threadIdx.x >> 5;
    const int lane = threadIdx.x & 31;
    const int nPanels = Ncols >> 5;
    const int bm = blockIdx.x / nPanels;
    const int bn = blockIdx.x % nPanels;

    // Wave 0 DMAs the 32 x Kpad B panel (contiguous rows of W^T) into LDS.
    if (wid == 0) {
        tdm_load_slab_f16(Wt + (size_t)(bn * 32) * ldw,
                          (unsigned)(uintptr_t)&ldsB[0], 32 * Kpad);
        __builtin_amdgcn_s_wait_tensorcnt(0);
    }
    __syncthreads();

    const int half = lane >> 4;
    const int l15  = lane & 15;
    const int row0 = bm * 256 + wid * 32;

    const _Float16* Ab0 = A + (size_t)(row0 + l15) * lda + half * 8;
    const _Float16* Ab1 = Ab0 + (size_t)16 * lda;
    const _Float16* Bb0 = ldsB + (size_t)l15 * ldw + half * 8;
    const _Float16* Bb1 = Bb0 + (size_t)16 * ldw;

    const int colb = bn * 32 + l15;
    const int rowb = row0 + half * 8;

    v8f acc[2][2];
    if (Cin) {
#pragma unroll
        for (int mi = 0; mi < 2; ++mi)
#pragma unroll
            for (int ni = 0; ni < 2; ++ni)
#pragma unroll
                for (int i = 0; i < 8; ++i)
                    acc[mi][ni][i] = Cin[(size_t)(rowb + mi * 16 + i) * Ncols + colb + ni * 16];
    } else {
#pragma unroll
        for (int mi = 0; mi < 2; ++mi)
#pragma unroll
            for (int ni = 0; ni < 2; ++ni)
#pragma unroll
                for (int i = 0; i < 8; ++i)
                    acc[mi][ni][i] = 0.f;
    }

    for (int k0 = 0; k0 < Kpad; k0 += 32) {
        __builtin_prefetch(Ab0 + k0 + 64, 0, 1);    // global_prefetch_b8 (A stream)
        v16h a0 = fragA(Ab0 + k0);
        v16h a1 = fragA(Ab1 + k0);
        v16h b0 = fragB(Bb0 + k0);
        v16h b1 = fragB(Bb1 + k0);
        acc[0][0] = __builtin_amdgcn_wmma_f32_16x16x32_f16(false, a0, false, b0, (short)0, acc[0][0], false, false);
        acc[0][1] = __builtin_amdgcn_wmma_f32_16x16x32_f16(false, a0, false, b1, (short)0, acc[0][1], false, false);
        acc[1][0] = __builtin_amdgcn_wmma_f32_16x16x32_f16(false, a1, false, b0, (short)0, acc[1][0], false, false);
        acc[1][1] = __builtin_amdgcn_wmma_f32_16x16x32_f16(false, a1, false, b1, (short)0, acc[1][1], false, false);
    }

#pragma unroll
    for (int mi = 0; mi < 2; ++mi)
#pragma unroll
        for (int ni = 0; ni < 2; ++ni) {
            int cc = colb + ni * 16;
            float bv = bias ? bias[cc] : 0.f;
#pragma unroll
            for (int i = 0; i < 8; ++i) {
                float v = acc[mi][ni][i] + bv;
                if (relu) v = fmaxf(v, 0.f);
                C[(size_t)(rowb + mi * 16 + i) * Ncols + cc] = v;
            }
        }
}

// ---------------------------------------------------------------------------
// Fallback WMMA GEMM (one 16x16 tile per wave) for small shapes.
// ---------------------------------------------------------------------------
__global__ void gemm_f16_wmma_kernel(const _Float16* __restrict__ A, int lda,
                                     const _Float16* __restrict__ Wt, int ldw,
                                     const float* __restrict__ bias,
                                     const float* __restrict__ Cin,
                                     float* __restrict__ C,
                                     int M, int Ncols, int Kpad, int relu) {
    int gw   = (int)((blockIdx.x * blockDim.x + threadIdx.x) >> 5);
    int lane = threadIdx.x & 31;
    int ntiles = Ncols >> 4;
    int total  = (M >> 4) * ntiles;
    if (gw >= total) return;
    int mt = gw / ntiles, nt = gw % ntiles;

    const _Float16* Abase = A  + (size_t)(mt * 16 + (lane & 15)) * lda + ((lane >> 4) * 8);
    const _Float16* Bbase = Wt + (size_t)(nt * 16 + (lane & 15)) * ldw + ((lane >> 4) * 8);

    int col  = nt * 16 + (lane & 15);
    int rowb = mt * 16 + ((lane >> 4) * 8);

    v8f acc;
    if (Cin) {
#pragma unroll
        for (int i = 0; i < 8; ++i) acc[i] = Cin[(size_t)(rowb + i) * Ncols + col];
    } else {
#pragma unroll
        for (int i = 0; i < 8; ++i) acc[i] = 0.f;
    }

    for (int k0 = 0; k0 < Kpad; k0 += 32) {
        __builtin_prefetch(Abase + k0 + 64, 0, 1);
        v16h a = fragA(Abase + k0);
        v16h b = fragB(Bbase + k0);
        acc = __builtin_amdgcn_wmma_f32_16x16x32_f16(false, a, false, b, (short)0, acc, false, false);
    }

    float bv = bias ? bias[col] : 0.f;
#pragma unroll
    for (int i = 0; i < 8; ++i) {
        float v = acc[i] + bv;
        if (relu) v = fmaxf(v, 0.f);
        C[(size_t)(rowb + i) * Ncols + col] = v;
    }
}

// ---------------------------------------------------------------------------
// Column-wise BatchNorm (training stats, biased var), f32 in -> f16 out x2
// ---------------------------------------------------------------------------
__global__ void bn_cols_kernel(const float* __restrict__ X, int M, int ncols,
                               const float* __restrict__ g, const float* __restrict__ be,
                               _Float16* __restrict__ Y1, int ldy1, int off1,
                               _Float16* __restrict__ Y2, int ldy2, int off2) {
    int c = blockIdx.x;
    int t = threadIdx.x;
    __shared__ float ssum[256], ssq[256];
    float s = 0.f, q = 0.f;
    for (int r = t; r < M; r += 256) {
        float v = X[(size_t)r * ncols + c]; s += v; q += v * v;
    }
    ssum[t] = s; ssq[t] = q; __syncthreads();
    for (int o = 128; o > 0; o >>= 1) {
        if (t < o) { ssum[t] += ssum[t + o]; ssq[t] += ssq[t + o]; }
        __syncthreads();
    }
    float mean = ssum[0] / M;
    float var  = ssq[0] / M - mean * mean;
    float inv  = rsqrtf(var + 1e-5f) * g[c];
    float beta = be[c];
    for (int r = t; r < M; r += 256) {
        float v = (X[(size_t)r * ncols + c] - mean) * inv + beta;
        _Float16 h = (_Float16)v;
        Y1[(size_t)r * ldy1 + off1 + c] = h;
        if (Y2) Y2[(size_t)r * ldy2 + off2 + c] = h;
    }
}

__global__ void cast_f16_kernel(const float* __restrict__ X, _Float16* __restrict__ Y, int n) {
    int t = blockIdx.x * blockDim.x + threadIdx.x;
    if (t < n) Y[t] = (_Float16)X[t];
}

// ---------------------------------------------------------------------------
// Fused GravNet: per-hit top-6 in 16-d space (per event), exp(-10 d^2)
// weighted mean+max aggregation of 64-d props -> 128-d agg (f16)
// ---------------------------------------------------------------------------
__global__ void gravnet_kernel(const float* __restrict__ S, const float* __restrict__ H,
                               _Float16* __restrict__ AGG) {
    const int wid  = threadIdx.x >> 5;
    const int lane = threadIdx.x & 31;
    const int row  = blockIdx.x * 8 + wid;
    const int base = (row >> 11) << 11;

    __shared__ float sd[8][KNN * 32];
    __shared__ int   si[8][KNN * 32];
    __shared__ float selD[8][KNN];
    __shared__ int   selI[8][KNN];

    float sn[SPACE_D]; float sqn = 0.f;
#pragma unroll
    for (int i = 0; i < SPACE_D; ++i) { sn[i] = S[(size_t)row * SPACE_D + i]; sqn += sn[i] * sn[i]; }

    float bd[KNN]; int bi[KNN];
#pragma unroll
    for (int j = 0; j < KNN; ++j) { bd[j] = 3.0e38f; bi[j] = base; }

    for (int m = lane; m < NN; m += 32) {
        const float* sm = S + (size_t)(base + m) * SPACE_D;
        float dot = 0.f, sqm = 0.f;
#pragma unroll
        for (int i = 0; i < SPACE_D; ++i) { float v = sm[i]; dot += sn[i] * v; sqm += v * v; }
        float d2 = sqn + sqm - 2.f * dot;
        if (d2 < 0.f) d2 = 0.f;
        if (d2 < bd[KNN - 1]) {
            int j = KNN - 1;
            while (j > 0 && bd[j - 1] > d2) { bd[j] = bd[j - 1]; bi[j] = bi[j - 1]; --j; }
            bd[j] = d2; bi[j] = base + m;
        }
    }
#pragma unroll
    for (int j = 0; j < KNN; ++j) { sd[wid][lane * KNN + j] = bd[j]; si[wid][lane * KNN + j] = bi[j]; }
    __syncthreads();
    if (lane == 0) {
        for (int j = 0; j < KNN; ++j) {
            float best = 3.0e38f; int bidx = base, bpos = 0;
            for (int t = 0; t < KNN * 32; ++t)
                if (sd[wid][t] < best) { best = sd[wid][t]; bidx = si[wid][t]; bpos = t; }
            sd[wid][bpos] = 3.1e38f;
            selD[wid][j] = best; selI[wid][j] = bidx;
        }
    }
    __syncthreads();

    float wgt[KNN]; int nb[KNN];
#pragma unroll
    for (int j = 0; j < KNN; ++j) { wgt[j] = __expf(-10.f * selD[wid][j]); nb[j] = selI[wid][j]; }

#pragma unroll
    for (int r = 0; r < 2; ++r) {
        int dme = lane + 32 * r;
        float sum = 0.f, mx = -3.0e38f;
#pragma unroll
        for (int j = 0; j < KNN; ++j) {
            float v = H[(size_t)nb[j] * PROP_D + dme] * wgt[j];
            sum += v; mx = fmaxf(mx, v);
        }
        AGG[(size_t)row * 128 + dme]      = (_Float16)(sum * (1.f / KNN));
        AGG[(size_t)row * 128 + 64 + dme] = (_Float16)mx;
    }
}

// ---------------------------------------------------------------------------
// Final pooling + dense head epilogue
// ---------------------------------------------------------------------------
__global__ void pool_kernel(const _Float16* __restrict__ C, _Float16* __restrict__ P) {
    int t = blockIdx.x * blockDim.x + threadIdx.x;
    if (t >= BB * 192) return;
    int e = t / 192, c = t % 192;
    const _Float16* p = C + (size_t)e * NN * 192 + c;
    float sum = 0.f, mn = 3.0e38f, mx = -3.0e38f;
    for (int n = 0; n < NN; ++n) {
        float v = (float)p[(size_t)n * 192];
        sum += v; mn = fminf(mn, v); mx = fmaxf(mx, v);
    }
    P[e * 768 + c]       = (_Float16)(sum * (1.f / NN));
    P[e * 768 + 192 + c] = (_Float16)mn;
    P[e * 768 + 384 + c] = (_Float16)mx;
    P[e * 768 + 576 + c] = (_Float16)sum;
}

__global__ void final_out_kernel(const _Float16* __restrict__ X, const float* __restrict__ Wout,
                                 const float* __restrict__ bout, float* __restrict__ out) {
    int b = threadIdx.x;
    if (b >= BB) return;
    float s = 0.f;
    for (int k = 0; k < HIDD; ++k) s += (float)X[b * HIDD + k] * Wout[k];
    out[b] = s + bout[0];
}

// ---------------------------------------------------------------------------
// Host side
// ---------------------------------------------------------------------------
struct BlkP { const float *W0,*W1,*W2,*Wh,*Wo1,*Wo2,*Ws,*b0,*b1,*b2,*be1,*be2,*be3,*bh,*bo2,*bs,*g1,*g2,*g3; };
struct DnsP { const float *W,*b,*be,*g; };

static void launch_prep(hipStream_t st, const float* W, _Float16* Wt, int K, int N, int Kpad) {
    int n = N * Kpad;
    prep_wt_kernel<<<(n + 255) / 256, 256, 0, st>>>(W, Wt, K, N, Kpad);
}
static void launch_gemm(hipStream_t st, const _Float16* A, int lda, const _Float16* Wt, int ldw,
                        const float* bias, const float* Cin, float* C,
                        int M, int Ncols, int Kpad, int relu) {
    if ((M % 256) == 0 && (Ncols % 32) == 0 && Kpad <= 128 && ldw == Kpad) {
        int tiles = (M / 256) * (Ncols / 32);
        gemm_f16_wmma_lds_kernel<<<tiles, 256, 0, st>>>(A, lda, Wt, ldw, bias, Cin, C, M, Ncols, Kpad, relu);
    } else {
        int tiles = (M / 16) * (Ncols / 16);
        gemm_f16_wmma_kernel<<<(tiles + 7) / 8, 256, 0, st>>>(A, lda, Wt, ldw, bias, Cin, C, M, Ncols, Kpad, relu);
    }
}

extern "C" void kernel_launch(void* const* d_in, const int* in_sizes, int n_in,
                              void* d_out, int out_size, void* d_ws, size_t ws_size,
                              hipStream_t stream) {
    auto F = [&](int i) { return (const float*)d_in[i]; };
    const float* x = F(0);            // (32768, 20)  [d_in[1] = batch, unused: fixed layout]

    int idx = 2;
    const float* Wout = F(idx++);
    BlkP blk[3];
    for (int b = 0; b < 3; ++b) {
        BlkP& p = blk[b];
        p.W0=F(idx++); p.W1=F(idx++); p.W2=F(idx++); p.Wh=F(idx++); p.Wo1=F(idx++);
        p.Wo2=F(idx++); p.Ws=F(idx++); p.b0=F(idx++); p.b1=F(idx++); p.b2=F(idx++);
        p.be1=F(idx++); p.be2=F(idx++); p.be3=F(idx++); p.bh=F(idx++); p.bo2=F(idx++);
        p.bs=F(idx++); p.g1=F(idx++); p.g2=F(idx++); p.g3=F(idx++);
    }
    const float* bout = F(idx++);
    DnsP dns[3];
    for (int i = 0; i < 3; ++i) { dns[i].W=F(idx++); dns[i].b=F(idx++); dns[i].be=F(idx++); dns[i].g=F(idx++); }

    char* w = (char*)d_ws; size_t off = 0;
    auto take = [&](size_t bytes) -> char* {
        char* p = w + off; off = (off + bytes + 255) & ~(size_t)255; return p;
    };
    float*    stats  = (float*)   take(BB * 60 * sizeof(float));
    _Float16* xf96   = (_Float16*)take((size_t)NH * 96  * 2);
    _Float16* wt     = (_Float16*)take((size_t)256 * 768 * 2);
    float*    c0     = (float*)   take((size_t)NH * 128 * 4);
    _Float16* a1     = (_Float16*)take((size_t)NH * 128 * 2);
    _Float16* x64    = (_Float16*)take((size_t)NH * 64  * 2);
    float*    sbuf   = (float*)   take((size_t)NH * SPACE_D * 4);
    float*    hbuf   = (float*)   take((size_t)NH * PROP_D  * 4);
    _Float16* agg    = (_Float16*)take((size_t)NH * 128 * 2);
    float*    c2     = (float*)   take((size_t)NH * 64  * 4);
    _Float16* concat = (_Float16*)take((size_t)NH * 192 * 2);
    _Float16* pooled = (_Float16*)take((size_t)BB * 768 * 2);
    float*    dc     = (float*)   take((size_t)BB * 256 * 4);
    _Float16* da     = (_Float16*)take((size_t)BB * 256 * 2);
    (void)ws_size; (void)in_sizes; (void)n_in; (void)out_size;

    in_stats_kernel<<<BB, 32, 0, stream>>>(x, stats);
    build_x96_kernel<<<(NH * 96 + 255) / 256, 256, 0, stream>>>(x, stats, xf96);

    for (int b = 0; b < 3; ++b) {
        const BlkP& p = blk[b];
        const _Float16* A0  = (b == 0) ? xf96 : x64;
        int             ld0 = (b == 0) ? 96 : 64;
        int             Kd0 = (b == 0) ? 80 : 64;

        launch_prep(stream, p.W0, wt, Kd0, PREHID, ld0);
        launch_gemm(stream, A0, ld0, wt, ld0, p.b0, nullptr, c0, NH, PREHID, ld0, 1);
        bn_cols_kernel<<<PREHID, 256, 0, stream>>>(c0, NH, PREHID, p.g1, p.be1, a1, PREHID, 0, nullptr, 0, 0);

        launch_prep(stream, p.W1, wt, PREHID, PREHID, PREHID);
        launch_gemm(stream, a1, PREHID, wt, PREHID, p.b1, nullptr, c0, NH, PREHID, PREHID, 1);
        bn_cols_kernel<<<PREHID, 256, 0, stream>>>(c0, NH, PREHID, p.g2, p.be2, a1, PREHID, 0, nullptr, 0, 0);

        launch_prep(stream, p.W2, wt, PREHID, GRAV, PREHID);
        launch_gemm(stream, a1, PREHID, wt, PREHID, p.b2, nullptr, c0, NH, GRAV, PREHID, 0);
        cast_f16_kernel<<<(NH * GRAV + 255) / 256, 256, 0, stream>>>(c0, x64, NH * GRAV);

        launch_prep(stream, p.Ws, wt, GRAV, SPACE_D, GRAV);
        launch_gemm(stream, x64, GRAV, wt, GRAV, p.bs, nullptr, sbuf, NH, SPACE_D, GRAV, 0);
        launch_prep(stream, p.Wh, wt, GRAV, PROP_D, GRAV);
        launch_gemm(stream, x64, GRAV, wt, GRAV, p.bh, nullptr, hbuf, NH, PROP_D, GRAV, 0);

        gravnet_kernel<<<NH / 8, 256, 0, stream>>>(sbuf, hbuf, agg);

        launch_prep(stream, p.Wo1, wt, GRAV, GRAV, GRAV);
        launch_gemm(stream, x64, GRAV, wt, GRAV, nullptr, nullptr, c2, NH, GRAV, GRAV, 0);
        launch_prep(stream, p.Wo2, wt, 2 * PROP_D, GRAV, 2 * PROP_D);
        launch_gemm(stream, agg, 128, wt, 128, p.bo2, c2, c2, NH, GRAV, 128, 0);

        bn_cols_kernel<<<GRAV, 256, 0, stream>>>(c2, NH, GRAV, p.g3, p.be3,
                                                 x64, GRAV, 0, concat, 192, 64 * b);
    }

    pool_kernel<<<(BB * 192 + 255) / 256, 256, 0, stream>>>(concat, pooled);

    int din = 768;
    const _Float16* Ad = pooled;
    for (int i = 0; i < 3; ++i) {
        launch_prep(stream, dns[i].W, wt, din, HIDD, din);
        launch_gemm(stream, Ad, din, wt, din, dns[i].b, nullptr, dc, BB, HIDD, din, 1);
        bn_cols_kernel<<<HIDD, 256, 0, stream>>>(dc, BB, HIDD, dns[i].g, dns[i].be, da, HIDD, 0, nullptr, 0, 0);
        Ad = da; din = HIDD;
    }
    final_out_kernel<<<1, 32, 0, stream>>>(da, Wout, bout, (float*)d_out);
}
